// Attention_9259949490680
// MI455X (gfx1250) — matrix operand order repaired and checked by reading, not compiled, not run on hardware
//
#include <hip/hip_runtime.h>

// ---------------------------------------------------------------------------
// Fused multi-head attention (softmax over HEAD axis) for MI455X / gfx1250.
// All heavy math through v_wmma_f32_16x16x32_bf16 (f32 accumulate).
// B=8, L=1024, D_MODEL=1024, H=16, Dh=64.
//
// B-operands (K^T, V, Wfc^T) are stored in *fragment-packed* order by the
// producer kernels: for each 32x16 WMMA B-tile, lane l's 16 bf16 values are
// contiguous at tile + l*16.  A B-fragment load is then one 32-byte
// per-lane vector load (2x global_load_b128, fully coalesced) instead of
// 16 strided b16 loads.
// ---------------------------------------------------------------------------

#define B_    8
#define L_    1024
#define DM_   1024
#define H_    16
#define DH_   64

typedef __attribute__((ext_vector_type(16))) __bf16 v16bf;
typedef __attribute__((ext_vector_type(8)))  float  v8f;

__device__ static inline v8f wmma_bf16(v16bf a, v16bf b, v8f c) {
  // (neg_a, A, neg_b, B, c_mod, C, reuse_a, reuse_b)
  return __builtin_amdgcn_wmma_f32_16x16x32_bf16(false, a, false, b, (short)0, c,
                                                 false, false);
}

// A-fragment (16x32, MxK) from row-major bf16, row stride lda (elements).
// Layout: lane m=lane&15; lane<16 holds K in {0..15}, lane>=16 {16..31}.
// Per lane this is two contiguous 16-byte runs -> compiler merges to b128 loads.
__device__ static inline v16bf load_a_frag(const __bf16* base, int lda, int lane) {
  int m  = lane & 15;
  int k0 = (lane >> 4) * 16;
  const __bf16* row = base + (size_t)m * lda;
  v16bf a;
#pragma unroll
  for (int i = 0; i < 8; ++i) {
    a[i]     = row[k0 + i];
    a[8 + i] = row[8 + k0 + i];
  }
  return a;
}

// Same A-fragment but sourced from fp32 memory (convert to bf16 on load).
__device__ static inline v16bf load_a_frag_f32(const float* base, int lda, int lane) {
  int m  = lane & 15;
  int k0 = (lane >> 4) * 16;
  const float* row = base + (size_t)m * lda;
  v16bf a;
#pragma unroll
  for (int i = 0; i < 8; ++i) {
    a[i]     = (__bf16)row[k0 + i];
    a[8 + i] = (__bf16)row[8 + k0 + i];
  }
  return a;
}

// Fragment-packed B tile: lane's 16 bf16 contiguous at tile + lane*16.
__device__ static inline v16bf load_b_packed(const __bf16* tile, int lane) {
  return *(const v16bf*)(tile + lane * 16);
}

// B-fragment for x @ W.T: B[k][n] = W[n][k], W fp32 row-major [64][64].
// Per lane this is 16 *contiguous* floats of row n starting at kb.
__device__ static inline v16bf load_b_frag_wt(const float* W, int ldw,
                                              int kchunk, int n0, int lane) {
  int n  = n0 + (lane & 15);
  int kb = kchunk + (lane >> 4) * 16;
  const float* row = W + (size_t)n * ldw + kb;
  v16bf b;
#pragma unroll
  for (int e = 0; e < 16; ++e) b[e] = (__bf16)row[e];
  return b;
}

// B-tile packing index: tile covers K rows kc*32..+31, N cols nt*16..+15.
//   lane = (n&15) + 16*((k>>4)&1), element = k&15.
// -> offset within tile = lane*16 + e  (512 bf16 per tile).

// ---------------------------------------------------------------------------
// K0: Wfc [1024 out][1024 in] fp32 -> WfF fragment-packed bf16.
// B[k][n] = Wfc[n][k]; tiles indexed [kc32:32][nt:64], 512 bf16 each.
// ---------------------------------------------------------------------------
__global__ void wfct_kernel(const float* __restrict__ Wfc, __bf16* __restrict__ WfF) {
  int idx = blockIdx.x * 256 + threadIdx.x;          // 1M elements
  int n = idx >> 10;                                 // consecutive threads vary k
  int k = idx & 1023;                                //  -> coalesced source reads
  int kc = k >> 5, nt = n >> 4;
  int lane = (n & 15) + ((k >> 4) & 1) * 16;
  int e    = k & 15;
  WfF[((size_t)(kc * 64 + nt) * 32 + lane) * 16 + e] = (__bf16)Wfc[(size_t)n * DM_ + k];
}

// ---------------------------------------------------------------------------
// K1: per-head-dim projections. One wave per (ltile, b*h, which).
//   Qp  [b][h][l][d]              bf16  (A-ready, row-major in d)
//   KpF [b][h][kt16:64][dc:2][512] bf16  (score B-tiles, fragment-packed)
//   VpF [b][h][kc32:32][dt:4][512] bf16  (PV B-tiles, fragment-packed)
// ---------------------------------------------------------------------------
__global__ void proj_kernel(const float* __restrict__ q, const float* __restrict__ k,
                            const float* __restrict__ v,
                            const float* __restrict__ Wq, const float* __restrict__ bq,
                            const float* __restrict__ Wk, const float* __restrict__ bk,
                            const float* __restrict__ Wv, const float* __restrict__ bv,
                            __bf16* __restrict__ Qp, __bf16* __restrict__ KpF,
                            __bf16* __restrict__ VpF) {
  int lane  = threadIdx.x;
  int which = blockIdx.z;                 // 0=Q 1=K 2=V (uniform -> EXEC stays full)
  int bh    = blockIdx.y;
  int b     = bh >> 4, h = bh & 15;
  int bx    = blockIdx.x;
  int l0    = bx * 16;

  const float* X;   const float* W;   const float* bias;
  if (which == 0)      { X = q; W = Wq; bias = bq; }
  else if (which == 1) { X = k; W = Wk; bias = bk; }
  else                 { X = v; W = Wv; bias = bv; }

  const float* Arow = X + ((size_t)b * L_ + l0) * DM_ + h * DH_;
  v16bf a0 = load_a_frag_f32(Arow, DM_, lane);        // d = 0..31
  v16bf a1 = load_a_frag_f32(Arow + 32, DM_, lane);   // d = 32..63

  v8f acc[4] = {};
#pragma unroll
  for (int nt = 0; nt < 4; ++nt) {
    v16bf b0 = load_b_frag_wt(W, DH_, 0,  nt * 16, lane);
    v16bf b1 = load_b_frag_wt(W, DH_, 32, nt * 16, lane);
    acc[nt] = wmma_bf16(a0, b0, acc[nt]);
    acc[nt] = wmma_bf16(a1, b1, acc[nt]);
  }

  int n = lane & 15, mb = (lane >> 4) * 8;
  size_t headbase = ((size_t)b * H_ + h);
  __bf16* KF = KpF + headbase * (64 * 2 * 512);
  __bf16* VF = VpF + headbase * (32 * 4 * 512);
#pragma unroll
  for (int nt = 0; nt < 4; ++nt) {
    int dp = nt * 16 + n;                // output channel d'
    float bval = bias[dp];
#pragma unroll
    for (int j = 0; j < 8; ++j) {
      float val = acc[nt][j] + bval;
      int m = mb + j;                    // row l0+m
      if (which == 0) {
        Qp[(headbase * L_ + (l0 + m)) * DH_ + dp] = (__bf16)val;
      } else if (which == 1) {
        // score B-tile: K dim = d (dp), N = kpos (l0+m)
        int kt16  = bx;                  // (l0+m)>>4
        int dc    = nt >> 1;             // dp>>5
        int lt    = m + 16 * (nt & 1);   // (kpos&15) + 16*((dp>>4)&1)
        int e     = n;                   // dp&15
        KF[((size_t)(kt16 * 2 + dc) * 32 + lt) * 16 + e] = (__bf16)val;
      } else {
        // PV B-tile: K dim = kpos (l0+m), N = d (dp)
        int kc32  = bx >> 1;             // (l0+m)>>5
        int dt    = nt;                  // dp>>4
        int lt    = n + 16 * (bx & 1);   // (dp&15) + 16*((kpos>>4)&1)
        int e     = m;                   // kpos&15
        VF[((size_t)(kc32 * 4 + dt) * 32 + lt) * 16 + e] = (__bf16)val;
      }
    }
  }
}

// ---------------------------------------------------------------------------
// K2: fused scores + head-axis softmax + P·V.
// One block = 16 waves = 16 heads, for one (b, 16-row q-tile).
// k processed in chunks of 32; score tiles exchanged through LDS so the
// softmax (over the 16 heads) is done elementwise by all 512 threads.
// Next iteration's K/V fragment tiles are prefetched (global_prefetch_b8).
// Output: Obuf [b][l][h*64+d] bf16 (row-major, feeds the FC as A-tiles).
// ---------------------------------------------------------------------------
__global__ void __launch_bounds__(512)
attn_kernel(const __bf16* __restrict__ Qp, const __bf16* __restrict__ KpF,
            const __bf16* __restrict__ VpF, __bf16* __restrict__ Obuf) {
  __shared__ float  Sbuf[H_][16][32];   // 32 KB: raw scores, [h][q][k]
  __shared__ __bf16 Pbuf[H_][16][32];   // 16 KB: softmaxed probs (bf16)

  int tid  = threadIdx.x;
  int lane = tid & 31;
  int h    = tid >> 5;
  int b    = blockIdx.y;
  int q0   = blockIdx.x * 16;

  const __bf16* Qh = Qp  + (((size_t)b * H_ + h) * L_ + q0) * DH_;
  const __bf16* KF = KpF + ((size_t)b * H_ + h) * (64 * 2 * 512);
  const __bf16* VF = VpF + ((size_t)b * H_ + h) * (32 * 4 * 512);

  // Q fragments for this head's q-tile live in registers the whole kernel.
  v16bf aq0 = load_a_frag(Qh, DH_, lane);        // d 0..31
  v16bf aq1 = load_a_frag(Qh + 32, DH_, lane);   // d 32..63

  v8f oacc[4] = {};                              // 16q x 64d output accumulators
  const float scale = 0.125f;                    // 1/sqrt(64)
  int n = lane & 15, mb = (lane >> 4) * 8;

  for (int kt = 0; kt < L_ / 32; ++kt) {
    // ---- prefetch next iteration's K/V fragment tiles (per-lane 32B) ----
    if (kt + 1 < L_ / 32) {
      int kt2 = kt + 1;
#pragma unroll
      for (int kn = 0; kn < 2; ++kn) {
        __builtin_prefetch(KF + ((size_t)(kt2 * 2 + kn) * 2 + 0) * 512 + lane * 16, 0, 1);
        __builtin_prefetch(KF + ((size_t)(kt2 * 2 + kn) * 2 + 1) * 512 + lane * 16, 0, 1);
      }
#pragma unroll
      for (int dt = 0; dt < 4; ++dt)
        __builtin_prefetch(VF + (size_t)(kt2 * 4 + dt) * 512 + lane * 16, 0, 1);
    }

    // ---- scores: two 16x16 subtiles (k columns kt*32 .. kt*32+31) ----
#pragma unroll
    for (int kn = 0; kn < 2; ++kn) {
      v16bf bk0 = load_b_packed(KF + ((size_t)(kt * 2 + kn) * 2 + 0) * 512, lane);
      v16bf bk1 = load_b_packed(KF + ((size_t)(kt * 2 + kn) * 2 + 1) * 512, lane);
      v8f s = {};
      s = wmma_bf16(aq0, bk0, s);
      s = wmma_bf16(aq1, bk1, s);
#pragma unroll
      for (int j = 0; j < 8; ++j)
        Sbuf[h][mb + j][kn * 16 + n] = s[j] * scale;
    }
    __syncthreads();

    // ---- softmax across the 16 heads: one (q,k) point per thread ----
    {
      int m  = tid >> 5;         // 0..15
      int kk = tid & 31;         // 0..31
      float mx = -3.0e38f;
#pragma unroll
      for (int hh = 0; hh < H_; ++hh) mx = fmaxf(mx, Sbuf[hh][m][kk]);
      float e[H_];
      float sum = 0.0f;
#pragma unroll
      for (int hh = 0; hh < H_; ++hh) {
        e[hh] = __expf(Sbuf[hh][m][kk] - mx);
        sum += e[hh];
      }
      float inv = 1.0f / sum;
#pragma unroll
      for (int hh = 0; hh < H_; ++hh)
        Pbuf[hh][m][kk] = (__bf16)(e[hh] * inv);
    }
    __syncthreads();

    // ---- P·V: A = P_h (16x32), B = V chunk (32 x 64), 4 d-subtiles ----
    v16bf ap = load_a_frag(&Pbuf[h][0][0], 32, lane);
#pragma unroll
    for (int dt = 0; dt < 4; ++dt) {
      v16bf bv = load_b_packed(VF + (size_t)(kt * 4 + dt) * 512, lane);
      oacc[dt] = wmma_bf16(ap, bv, oacc[dt]);
    }
    __syncthreads();   // protect Pbuf/Sbuf before next iteration overwrites
  }

  // ---- store: Obuf[b][q0+m][h*64 + d] ----
  __bf16* Orow = Obuf + ((size_t)b * L_ + q0) * DM_ + h * DH_;
#pragma unroll
  for (int dt = 0; dt < 4; ++dt) {
#pragma unroll
    for (int j = 0; j < 8; ++j)
      Orow[(size_t)(mb + j) * DM_ + dt * 16 + n] = (__bf16)oacc[dt][j];
  }
}

// ---------------------------------------------------------------------------
// K3: final FC  out = Obuf @ Wfc.T + bfc   ([8192 x 1024] x [1024 x 1024]).
// One wave per 16x64 output tile; K-loop of 32 chunks -> 128 WMMAs per wave.
// B-tiles come fragment-packed from WfF.
// ---------------------------------------------------------------------------
__global__ void fc_kernel(const __bf16* __restrict__ Obuf,
                          const __bf16* __restrict__ WfF,
                          const float* __restrict__ bfc, float* __restrict__ out) {
  int lane = threadIdx.x;
  int row0 = blockIdx.x * 16;        // over B*L = 8192 rows
  int n0   = blockIdx.y * 64;        // over 1024 outputs
  int ntb  = n0 >> 4;                // first of 4 n-tiles

  const __bf16* A = Obuf + (size_t)row0 * DM_;
  v8f acc[4] = {};
  for (int kc = 0; kc < DM_ / 32; ++kc) {
    // prefetch next K-chunk's A row data and B tiles
    if (kc + 1 < DM_ / 32) {
      __builtin_prefetch(A + (kc + 1) * 32 + (lane & 15) * DM_, 0, 1);
#pragma unroll
      for (int dt = 0; dt < 4; ++dt)
        __builtin_prefetch(WfF + (size_t)((kc + 1) * 64 + ntb + dt) * 512 + lane * 16, 0, 1);
    }
    v16bf a = load_a_frag(A + kc * 32, DM_, lane);
#pragma unroll
    for (int dt = 0; dt < 4; ++dt) {
      v16bf bfrag = load_b_packed(WfF + (size_t)(kc * 64 + ntb + dt) * 512, lane);
      acc[dt] = wmma_bf16(a, bfrag, acc[dt]);
    }
  }

  int n = lane & 15, mb = (lane >> 4) * 8;
#pragma unroll
  for (int dt = 0; dt < 4; ++dt) {
    float bv = bfc[n0 + dt * 16 + n];
#pragma unroll
    for (int j = 0; j < 8; ++j)
      out[(size_t)(row0 + mb + j) * DM_ + n0 + dt * 16 + n] = acc[dt][j] + bv;
  }
}

// ---------------------------------------------------------------------------
// Host launch.  Inputs (fp32): query,key,value,Wq,bq,Wk,bk,Wv,bv,Wfc,bfc.
// Workspace layout (bytes):
//   [ 0,16M)  Qp     [16M,32M) KpF    [32M,48M) VpF
//   [48M,64M) Obuf   [64M,66M) WfF
// ---------------------------------------------------------------------------
extern "C" void kernel_launch(void* const* d_in, const int* in_sizes, int n_in,
                              void* d_out, int out_size, void* d_ws, size_t ws_size,
                              hipStream_t stream) {
  const float* q   = (const float*)d_in[0];
  const float* k   = (const float*)d_in[1];
  const float* v   = (const float*)d_in[2];
  const float* Wq  = (const float*)d_in[3];
  const float* bq  = (const float*)d_in[4];
  const float* Wk  = (const float*)d_in[5];
  const float* bk  = (const float*)d_in[6];
  const float* Wv  = (const float*)d_in[7];
  const float* bv  = (const float*)d_in[8];
  const float* Wfc = (const float*)d_in[9];
  const float* bfc = (const float*)d_in[10];
  float* out = (float*)d_out;

  char* ws = (char*)d_ws;
  const size_t MB = 1024 * 1024;
  __bf16* Qp   = (__bf16*)(ws + 0 * MB);
  __bf16* KpF  = (__bf16*)(ws + 16 * MB);
  __bf16* VpF  = (__bf16*)(ws + 32 * MB);
  __bf16* Obuf = (__bf16*)(ws + 48 * MB);
  __bf16* WfF  = (__bf16*)(ws + 64 * MB);

  // K0: pack+convert Wfc (1M elements)
  wfct_kernel<<<dim3(DM_ * DM_ / 256), dim3(256), 0, stream>>>(Wfc, WfF);

  // K1: projections. grid = (L/16, B*H, 3), 1 wave per block.
  proj_kernel<<<dim3(L_ / 16, B_ * H_, 3), dim3(32), 0, stream>>>(
      q, k, v, Wq, bq, Wk, bk, Wv, bv, Qp, KpF, VpF);

  // K2: fused attention. grid = (L/16, B), 16 waves per block.
  attn_kernel<<<dim3(L_ / 16, B_), dim3(512), 0, stream>>>(Qp, KpF, VpF, Obuf);

  // K3: FC. grid = (B*L/16, DM/64), 1 wave per block.
  fc_kernel<<<dim3(B_ * L_ / 16, DM_ / 64), dim3(32), 0, stream>>>(
      Obuf, WfF, bfc, out);
}
